// AttentiveCNW_3848290697830
// MI455X (gfx1250) — matrix operand order repaired and checked by reading, not compiled, not run on hardware
//
#include <hip/hip_runtime.h>
#include <hip/hip_bf16.h>
#include <math.h>

#define B_   2
#define C_   256
#define HS   64
#define WS   64
#define NQ_  (HS*WS)      // 4096
#define NH_  8
#define NP_  8
#define DH_  (C_/NH_)     // 32
#define FFN_ 256
#define M_   (B_*NQ_)     // 8192 rows for every GEMM

typedef __attribute__((ext_vector_type(16))) __bf16 v16bf;
typedef __attribute__((ext_vector_type(8)))  float  v8f;
typedef __attribute__((ext_vector_type(4)))  int    v4i;

// ---------------------------------------------------------------------------
// CDNA5 async global->LDS copy (ASYNCcnt path), with portable fallback
// builtin signature (from clang): void(v4i AS(1)*, v4i AS(3)*, imm int, imm int)
// ---------------------------------------------------------------------------
#if defined(__gfx1250__) && __has_builtin(__builtin_amdgcn_global_load_async_to_lds_b128)
#define HAVE_ASYNC_LDS 1
#else
#define HAVE_ASYNC_LDS 0
#endif

__device__ __forceinline__ void copy16_g2l(const void* g, void* l) {
#if HAVE_ASYNC_LDS
  typedef __attribute__((address_space(1))) v4i*  gvec_p;
  typedef __attribute__((address_space(3))) v4i*  lvec_p;
  typedef __attribute__((address_space(1))) void* gvoid_p;
  typedef __attribute__((address_space(3))) void* lvoid_p;
  __builtin_amdgcn_global_load_async_to_lds_b128(
      (gvec_p)(gvoid_p)const_cast<void*>(g),
      (lvec_p)(lvoid_p)l,
      0, 0);
#else
  *(uint4*)l = *(const uint4*)g;
#endif
}
__device__ __forceinline__ void wait_async_le6() {
#if HAVE_ASYNC_LDS
  asm volatile("s_wait_asynccnt 0x6" ::: "memory");
#endif
}
__device__ __forceinline__ void wait_async_zero() {
#if HAVE_ASYNC_LDS
  asm volatile("s_wait_asynccnt 0x0" ::: "memory");
#endif
}

// ---------------------------------------------------------------------------
// Elementwise / prep kernels
// ---------------------------------------------------------------------------

// fused[b,q,c] = softmax(fuse_logit,axis=0) weighted sum of feat0/feat1 (NCHW -> NHWC)
__global__ void k_fuse(const float* __restrict__ f0, const float* __restrict__ f1,
                       const float* __restrict__ logit, float* __restrict__ fused) {
  int idx = blockIdx.x * blockDim.x + threadIdx.x;
  if (idx >= B_ * NQ_ * C_) return;
  int c = idx % C_;
  int q = (idx / C_) % NQ_;
  int b = idx / (C_ * NQ_);
  float l0 = logit[c], l1 = logit[C_ + c];
  float m  = fmaxf(l0, l1);
  float e0 = __expf(l0 - m), e1 = __expf(l1 - m);
  float w0 = e0 / (e0 + e1);
  float v0 = f0[((size_t)b * C_ + c) * NQ_ + q];
  float v1 = f1[((size_t)b * C_ + c) * NQ_ + q];
  fused[idx] = v0 * w0 + v1 * (1.f - w0);
}

// sine positional embedding [NQ, C]
__global__ void k_pos(float* __restrict__ pos) {
  int idx = blockIdx.x * blockDim.x + threadIdx.x;
  if (idx >= NQ_ * C_) return;
  int c = idx % C_, q = idx / C_;
  const float eps = 1e-6f, scale = 6.28318530717958647692f;
  int f; float coord;
  if (c < C_ / 2) { f = c;          coord = (float)(q / WS + 1) / ((float)HS + eps) * scale; }
  else            { f = c - C_ / 2; coord = (float)(q % WS + 1) / ((float)WS + eps) * scale; }
  float dimt = __powf(10000.f, (float)(2 * (f / 2)) / (float)(C_ / 2));
  float v = coord / dimt;
  pos[idx] = (f & 1) ? __cosf(v) : __sinf(v);
}

// split query_embed into qe[NQ,C] and tgt[B,NQ,C] (tgt broadcast over B)
__global__ void k_init_qe_tgt(const float* __restrict__ query_embed,
                              float* __restrict__ qe_buf, float* __restrict__ tgt) {
  int idx = blockIdx.x * blockDim.x + threadIdx.x;
  if (idx >= B_ * NQ_ * C_) return;
  int c = idx % C_;
  int q = (idx / C_) % NQ_;
  int b = idx / (C_ * NQ_);
  if (b == 0) qe_buf[q * C_ + c] = query_embed[(size_t)q * 2 * C_ + c];
  tgt[idx] = query_embed[(size_t)q * 2 * C_ + C_ + c];
}

// ref[q,0:2] = sigmoid(qe[q,:] @ ref_w + ref_b)
__global__ void k_refpts(const float* __restrict__ query_embed,
                         const float* __restrict__ rw, const float* __restrict__ rb,
                         float* __restrict__ ref) {
  int q = blockIdx.x * blockDim.x + threadIdx.x;
  if (q >= NQ_) return;
  const float* row = query_embed + (size_t)q * 2 * C_;
  float s0 = 0.f, s1 = 0.f;
  for (int k = 0; k < C_; ++k) {
    float x = row[k];
    s0 += x * rw[k * 2 + 0];
    s1 += x * rw[k * 2 + 1];
  }
  ref[q * 2 + 0] = 1.f / (1.f + __expf(-(s0 + rb[0])));
  ref[q * 2 + 1] = 1.f / (1.f + __expf(-(s1 + rb[1])));
}

// transpose+convert weight [K,N] f32 -> [N,K] bf16 for WMMA B operand
__global__ void k_wconv(const float* __restrict__ in, __bf16* __restrict__ out, int K, int N) {
  int idx = blockIdx.x * blockDim.x + threadIdx.x;
  if (idx >= K * N) return;
  int k = idx / N, n = idx % N;
  out[(size_t)n * K + k] = (__bf16)in[idx];
}

// ---------------------------------------------------------------------------
// LayerNorm (+ optional additive broadcast) — one wave32 per row of 256
// DO_LN=false => y = x (+ add) (used for self-attn query = tgt + pos)
// ---------------------------------------------------------------------------
template <bool DO_LN>
__global__ void k_ln_add(const float* __restrict__ x, const float* __restrict__ g,
                         const float* __restrict__ beta, const float* __restrict__ add,
                         float* __restrict__ y) {
  int wid  = (blockIdx.x * blockDim.x + threadIdx.x) >> 5;
  int lane = threadIdx.x & 31;
  if (wid >= M_) return;
  int q = wid & (NQ_ - 1);
  const float* xr = x + (size_t)wid * C_;
  float v[8];
  float s = 0.f;
#pragma unroll
  for (int i = 0; i < 8; ++i) { v[i] = xr[lane + 32 * i]; s += v[i]; }
  float mean = 0.f, rstd = 0.f;
  if (DO_LN) {
#pragma unroll
    for (int o = 16; o >= 1; o >>= 1) s += __shfl_xor(s, o, 32);
    mean = s * (1.f / C_);
    float vs = 0.f;
#pragma unroll
    for (int i = 0; i < 8; ++i) { float d = v[i] - mean; vs += d * d; }
#pragma unroll
    for (int o = 16; o >= 1; o >>= 1) vs += __shfl_xor(vs, o, 32);
    rstd = rsqrtf(vs * (1.f / C_) + 1e-5f);
  }
  float* yr = y + (size_t)wid * C_;
#pragma unroll
  for (int i = 0; i < 8; ++i) {
    int c = lane + 32 * i;
    float o = v[i];
    if (DO_LN) o = (o - mean) * rstd * g[c] + beta[c];
    if (add)   o += add[(size_t)q * C_ + c];
    yr[c] = o;
  }
}

// ---------------------------------------------------------------------------
// GEMM: Y[M,N] = act(X[M,K] @ W[K,N] + bias) (+ residual R)
// X f32 row-major (bf16-converted on fragment read), Wt bf16 [N,K].
// Block = 128 threads (4 waves) -> 64x64 output tile. K staged through
// double-buffered LDS via async global->LDS b128 copies (ASYNCcnt),
// 4x v_wmma_f32_16x16x32_bf16 per wave per K-slice.
// ---------------------------------------------------------------------------
#define BM 64
#define BN 64
#define BK 32

template <bool RELU, bool RESID>
__global__ __launch_bounds__(128)
void k_gemm(const float* __restrict__ X, const __bf16* __restrict__ Wt,
            const float* __restrict__ bias, const float* R,
            float* Y, int N, int K) {
  __shared__ float  As[2][BM * BK];   // 2 x 8 KB
  __shared__ __bf16 Bs[2][BN * BK];   // 2 x 4 KB

  const int tid  = threadIdx.x;
  const int wave = tid >> 5;
  const int lane = tid & 31;
  const int half = lane >> 4;
  const int r    = lane & 15;
  const int m_base = blockIdx.y * BM;
  const int n_base = blockIdx.x * BN;

  v8f acc[4];
#pragma unroll
  for (int j = 0; j < 4; ++j) acc[j] = (v8f){0.f, 0.f, 0.f, 0.f, 0.f, 0.f, 0.f, 0.f};

  // stage one BMxBK A slice (f32) + BNxBK B slice (bf16): 6 b128 copies/thread
  auto copy_tile = [&](int k0, int buf) {
#pragma unroll
    for (int s = 0; s < 4; ++s) {          // A: 512 x 16B chunks
      int ch   = tid + s * 128;
      int row  = ch >> 3;
      int koff = (ch & 7) * 4;             // f32 units
      copy16_g2l(X + (size_t)(m_base + row) * K + k0 + koff,
                 &As[buf][row * BK + koff]);
    }
#pragma unroll
    for (int s = 0; s < 2; ++s) {          // B: 256 x 16B chunks
      int ch   = tid + s * 128;
      int col  = ch >> 2;
      int koff = (ch & 3) * 8;             // bf16 units
      copy16_g2l(Wt + (size_t)(n_base + col) * K + k0 + koff,
                 &Bs[buf][col * BK + koff]);
    }
  };

  copy_tile(0, 0);
  const int KT = K >> 5;
  for (int kt = 0; kt < KT; ++kt) {
    const int buf = kt & 1;
    if (kt + 1 < KT) { copy_tile((kt + 1) << 5, buf ^ 1); wait_async_le6(); }
    else             { wait_async_zero(); }
    __syncthreads();

    // A fragment (16x32): lanes 0-15 hold K 0..15; lanes 16-31 K 16..31
    const float* Ar = &As[buf][(wave * 16 + r) * BK];
    const int ka = half ? 16 : 0;
    v16bf a;
#pragma unroll
    for (int i = 0; i < 8; ++i) {
      a[i]     = (__bf16)Ar[ka + i];
      a[8 + i] = (__bf16)Ar[ka + 8 + i];
    }
    const int kb = half ? 16 : 0;
#pragma unroll
    for (int j = 0; j < 4; ++j) {
      // B fragment (32x16): lane = column; lower half K 0..15, upper half K 16..31
      v16bf b = *(const v16bf*)&Bs[buf][(j * 16 + r) * BK + kb];
      acc[j] = __builtin_amdgcn_wmma_f32_16x16x32_bf16(false, a, false, b, (short)0,
                                                       acc[j], false, false);
    }
    __syncthreads();
  }

  // D layout: lanes 0-15: VGPR e -> M=e; lanes 16-31: VGPR e -> M=8+e; N = lane%16
#pragma unroll
  for (int j = 0; j < 4; ++j) {
    int col = n_base + j * 16 + r;
    float bv = bias[col];
#pragma unroll
    for (int e = 0; e < 8; ++e) {
      int row = m_base + wave * 16 + e + half * 8;
      float yv = acc[j][e] + bv;
      if (RELU)  yv = fmaxf(yv, 0.f);
      if (RESID) yv += R[(size_t)row * N + col];
      Y[(size_t)row * N + col] = yv;
    }
  }
}

// ---------------------------------------------------------------------------
// Deformable sampling: one wave32 per (b,q,h); lane = head channel (DH=32).
// Fuses the NP=8 softmax with the bilinear gather/weighted sum.
// ---------------------------------------------------------------------------
__global__ void k_msdeform(const float* __restrict__ value, const float* __restrict__ off,
                           const float* __restrict__ awl, const float* __restrict__ ref,
                           int pixel_mode, float* __restrict__ out) {
  int wid  = (blockIdx.x * blockDim.x + threadIdx.x) >> 5;
  int lane = threadIdx.x & 31;
  if (wid >= B_ * NQ_ * NH_) return;
  int h = wid % NH_;
  int q = (wid / NH_) % NQ_;
  int b = wid / (NH_ * NQ_);

  float rx, ry;
  if (pixel_mode) { rx = (float)(q % WS) + 0.5f; ry = (float)(q / WS) + 0.5f; }
  else            { rx = ref[q * 2 + 0];         ry = ref[q * 2 + 1]; }

  size_t obase = ((size_t)(b * NQ_ + q) * NH_ + h) * NP_;
  float lg[NP_];
  float mx = -3.0e38f;
#pragma unroll
  for (int p = 0; p < NP_; ++p) { lg[p] = awl[obase + p]; mx = fmaxf(mx, lg[p]); }
  float se = 0.f;
#pragma unroll
  for (int p = 0; p < NP_; ++p) { lg[p] = __expf(lg[p] - mx); se += lg[p]; }
  float inv = 1.f / se;

  const float* vb = value + (size_t)b * NQ_ * C_ + h * DH_ + lane;
  float acc = 0.f;
#pragma unroll
  for (int p = 0; p < NP_; ++p) {
    float ox = off[(obase + p) * 2 + 0];
    float oy = off[(obase + p) * 2 + 1];
    float px = (rx + ox * (1.f / WS)) * WS - 0.5f;
    float py = (ry + oy * (1.f / HS)) * HS - 0.5f;
    float fx = floorf(px), fy = floorf(py);
    float dx = px - fx, dy = py - fy;
    int x0 = (int)fx, y0 = (int)fy;
    float w   = lg[p] * inv;
    float w00 = (1.f - dx) * (1.f - dy) * w;
    float w10 = dx * (1.f - dy) * w;
    float w01 = (1.f - dx) * dy * w;
    float w11 = dx * dy * w;
    int x1 = x0 + 1, y1 = y0 + 1;
    if (x0 >= 0 && x0 < WS && y0 >= 0 && y0 < HS) acc += w00 * vb[(size_t)(y0 * WS + x0) * C_];
    if (x1 >= 0 && x1 < WS && y0 >= 0 && y0 < HS) acc += w10 * vb[(size_t)(y0 * WS + x1) * C_];
    if (x0 >= 0 && x0 < WS && y1 >= 0 && y1 < HS) acc += w01 * vb[(size_t)(y1 * WS + x0) * C_];
    if (x1 >= 0 && x1 < WS && y1 >= 0 && y1 < HS) acc += w11 * vb[(size_t)(y1 * WS + x1) * C_];
  }
  out[(size_t)(b * NQ_ + q) * C_ + h * DH_ + lane] = acc;
}

// ---------------------------------------------------------------------------
// Host orchestration
// ---------------------------------------------------------------------------
extern "C" void kernel_launch(void* const* d_in, const int* in_sizes, int n_in,
                              void* d_out, int out_size, void* d_ws, size_t ws_size,
                              hipStream_t stream) {
  (void)in_sizes; (void)n_in; (void)out_size; (void)ws_size;

  const float* feat0       = (const float*)d_in[0];
  const float* feat1       = (const float*)d_in[1];
  const float* query_embed = (const float*)d_in[2];
  const float* ref_w       = (const float*)d_in[3];
  const float* ref_b       = (const float*)d_in[4];
  const float* fuse_logit  = (const float*)d_in[5];
  const float* ln1_g       = (const float*)d_in[6];
  const float* ln1_b       = (const float*)d_in[7];
  const float* off_w       = (const float*)d_in[8];
  const float* off_bias    = (const float*)d_in[9];
  const float* aw_w        = (const float*)d_in[10];
  const float* aw_bias     = (const float*)d_in[11];
  const float* val_w       = (const float*)d_in[12];
  const float* val_bias    = (const float*)d_in[13];
  const float* out_w       = (const float*)d_in[14];
  const float* out_bias    = (const float*)d_in[15];
  const float* ln2_g       = (const float*)d_in[16];
  const float* ln2_b       = (const float*)d_in[17];
  const float* f1_w        = (const float*)d_in[18];
  const float* f1_bias     = (const float*)d_in[19];
  const float* f2_w        = (const float*)d_in[20];
  const float* f2_bias     = (const float*)d_in[21];

  // bump allocator on d_ws
  char* ws = (char*)d_ws;
  size_t cur = 0;
  auto alloc = [&](size_t bytes) -> char* {
    char* p = ws + cur;
    cur = (cur + bytes + 255) & ~(size_t)255;
    return p;
  };

  __bf16* offT[4]; __bf16* awT[4]; __bf16* valT[4]; __bf16* outT[4]; __bf16* f1T[4]; __bf16* f2T[4];
  for (int i = 0; i < 4; ++i) offT[i] = (__bf16*)alloc((size_t)C_ * 128 * 2);
  for (int i = 0; i < 4; ++i) awT[i]  = (__bf16*)alloc((size_t)C_ * 64 * 2);
  for (int i = 0; i < 4; ++i) valT[i] = (__bf16*)alloc((size_t)C_ * C_ * 2);
  for (int i = 0; i < 4; ++i) outT[i] = (__bf16*)alloc((size_t)C_ * C_ * 2);
  for (int i = 0; i < 4; ++i) f1T[i]  = (__bf16*)alloc((size_t)C_ * FFN_ * 2);
  for (int i = 0; i < 4; ++i) f2T[i]  = (__bf16*)alloc((size_t)FFN_ * C_ * 2);

  float* fused    = (float*)alloc((size_t)M_ * C_ * 4);
  float* pos      = (float*)alloc((size_t)NQ_ * C_ * 4);
  float* qe_buf   = (float*)alloc((size_t)NQ_ * C_ * 4);
  float* tgt      = (float*)alloc((size_t)M_ * C_ * 4);
  float* refbuf   = (float*)alloc((size_t)NQ_ * 2 * 4);
  float* t2q      = (float*)alloc((size_t)M_ * C_ * 4);
  float* offbuf   = (float*)alloc((size_t)M_ * 128 * 4);
  float* awlbuf   = (float*)alloc((size_t)M_ * 64 * 4);
  float* valuebuf = (float*)alloc((size_t)M_ * C_ * 4);
  float* sampled  = (float*)alloc((size_t)M_ * C_ * 4);
  float* hid      = (float*)alloc((size_t)M_ * FFN_ * 4);

  const int T = 256;
  // prep
  k_fuse<<<(B_ * NQ_ * C_ + T - 1) / T, T, 0, stream>>>(feat0, feat1, fuse_logit, fused);
  k_pos<<<(NQ_ * C_ + T - 1) / T, T, 0, stream>>>(pos);
  k_init_qe_tgt<<<(B_ * NQ_ * C_ + T - 1) / T, T, 0, stream>>>(query_embed, qe_buf, tgt);
  k_refpts<<<(NQ_ + T - 1) / T, T, 0, stream>>>(query_embed, ref_w, ref_b, refbuf);

  for (int i = 0; i < 4; ++i) {
    k_wconv<<<(C_ * 128 + T - 1) / T, T, 0, stream>>>(off_w + (size_t)i * C_ * 128, offT[i], C_, 128);
    k_wconv<<<(C_ * 64  + T - 1) / T, T, 0, stream>>>(aw_w  + (size_t)i * C_ * 64,  awT[i],  C_, 64);
    k_wconv<<<(C_ * C_  + T - 1) / T, T, 0, stream>>>(val_w + (size_t)i * C_ * C_,  valT[i], C_, C_);
    k_wconv<<<(C_ * C_  + T - 1) / T, T, 0, stream>>>(out_w + (size_t)i * C_ * C_,  outT[i], C_, C_);
    k_wconv<<<(C_ * FFN_ + T - 1) / T, T, 0, stream>>>(f1_w + (size_t)i * C_ * FFN_, f1T[i], C_, FFN_);
    k_wconv<<<(FFN_ * C_ + T - 1) / T, T, 0, stream>>>(f2_w + (size_t)i * FFN_ * C_, f2T[i], FFN_, C_);
  }

  auto gemm = [&](const float* X, const __bf16* Wt, const float* bias, const float* R,
                  float* Y, int N, int K, bool relu, bool resid) {
    dim3 grid(N / BN, M_ / BM);
    dim3 blk(128);
    if (relu && resid)      k_gemm<true,  true ><<<grid, blk, 0, stream>>>(X, Wt, bias, R, Y, N, K);
    else if (relu)          k_gemm<true,  false><<<grid, blk, 0, stream>>>(X, Wt, bias, nullptr, Y, N, K);
    else if (resid)         k_gemm<false, true ><<<grid, blk, 0, stream>>>(X, Wt, bias, R, Y, N, K);
    else                    k_gemm<false, false><<<grid, blk, 0, stream>>>(X, Wt, bias, nullptr, Y, N, K);
  };

  const int lnBlocks   = (M_ * 32 + T - 1) / T;            // one wave per row
  const int sampBlocks = (B_ * NQ_ * NH_ * 32 + T - 1) / T;

  for (int i = 0; i < 4; ++i) {
    bool cross = (i < 2);
    const float* src = cross ? fused : tgt;       // value-projection input

    // query input: cross -> LN1(tgt)+qe ; self -> tgt+pos (no LN in attn branch)
    if (cross)
      k_ln_add<true ><<<lnBlocks, T, 0, stream>>>(tgt, ln1_g + i * C_, ln1_b + i * C_, qe_buf, t2q);
    else
      k_ln_add<false><<<lnBlocks, T, 0, stream>>>(tgt, nullptr, nullptr, pos, t2q);

    gemm(t2q, offT[i], off_bias + (size_t)i * 128, nullptr, offbuf, 128, C_, false, false);
    gemm(t2q, awT[i],  aw_bias  + (size_t)i * 64,  nullptr, awlbuf, 64,  C_, false, false);
    gemm(src, valT[i], val_bias + (size_t)i * C_,  nullptr, valuebuf, C_, C_, false, false);

    k_msdeform<<<sampBlocks, T, 0, stream>>>(valuebuf, offbuf, awlbuf, refbuf,
                                             cross ? 0 : 1, sampled);

    // tgt = tgt + out_proj(sampled)
    gemm(sampled, outT[i], out_bias + (size_t)i * C_, tgt, tgt, C_, C_, false, true);

    // FFN: tgt = tgt + relu(relu(LN2(tgt)@f1+b1)@f2+b2)
    k_ln_add<true><<<lnBlocks, T, 0, stream>>>(tgt, ln2_g + i * C_, ln2_b + i * C_, nullptr, t2q);
    gemm(t2q, f1T[i], f1_bias + (size_t)i * FFN_, nullptr, hid, FFN_, C_, true, false);
    float* ydst = (i == 3) ? (float*)d_out : tgt; // final layer writes straight to output
    gemm(hid, f2T[i], f2_bias + (size_t)i * C_, tgt, ydst, C_, FFN_, true, true);
  }
}
